// QuantumNeuralNetwork_33818572489207
// MI455X (gfx1250) — compile-verified, matrix operand order checked by reading, not run on hardware
//
#include <hip/hip_runtime.h>
#include <hip/hip_bf16.h>

typedef __attribute__((ext_vector_type(16))) _Float16 v16h;
typedef __attribute__((ext_vector_type(8)))  float    v8f;

union V16H { v16h v; unsigned int u[8]; _Float16 e[16]; };

// ---------------------------------------------------------------------------
// Setup kernel (1 block, 32 threads): build the fixed 16x16 entangling unitary
// W from `weights`, fold the (-i)^popcount input phases into its columns,
// build G^T = (Z-sign pattern) x fc_w, pad fc_b. All exact f32 math; results
// stored as f16 in WMMA-friendly row-major layouts.
// ---------------------------------------------------------------------------
__global__ void qnn_setup(const float* __restrict__ weights,   // [4][4][3]
                          const float* __restrict__ fc_w,      // [10][4]
                          const float* __restrict__ fc_b,      // [10]
                          _Float16* __restrict__ wre,          // [16][16] (j,i)
                          _Float16* __restrict__ wim,          // [16][16]
                          _Float16* __restrict__ gt,           // [16][16] (c,j)
                          float* __restrict__ bpad)            // [16]
{
    const int tid = threadIdx.x;
    if (tid < 16) {
        const int i = tid;                      // basis-state column
        float cre[16], cim[16];
        #pragma unroll
        for (int j = 0; j < 16; ++j) { cre[j] = (j == i) ? 1.f : 0.f; cim[j] = 0.f; }

        for (int l = 0; l < 4; ++l) {
            // Rot(phi,theta,omega) on each wire (wire w <-> index bit 3-w, MSB = wire0)
            for (int w = 0; w < 4; ++w) {
                const float phi = weights[(l*4 + w)*3 + 0];
                const float th  = weights[(l*4 + w)*3 + 1];
                const float om  = weights[(l*4 + w)*3 + 2];
                const float ct  = cosf(0.5f*th), st = sinf(0.5f*th);
                const float sp  = 0.5f*(phi + om), sm = 0.5f*(phi - om);
                const float csp = cosf(sp), ssp = sinf(sp);
                const float csm = cosf(sm), ssm = sinf(sm);
                // U = [[a,b],[d,e]] : a=e^{-i sp}ct, b=-e^{+i sm}st, d=e^{-i sm}st, e=e^{+i sp}ct
                const float ar =  csp*ct, ai = -ssp*ct;
                const float br = -csm*st, bi = -ssm*st;
                const float dr =  csm*st, di = -ssm*st;
                const float er =  csp*ct, ei =  ssp*ct;
                const int tb = 1 << (3 - w);
                #pragma unroll
                for (int j0 = 0; j0 < 16; ++j0) {
                    if (j0 & tb) continue;
                    const int j1 = j0 | tb;
                    const float x0r = cre[j0], x0i = cim[j0];
                    const float x1r = cre[j1], x1i = cim[j1];
                    cre[j0] = ar*x0r - ai*x0i + br*x1r - bi*x1i;
                    cim[j0] = ar*x0i + ai*x0r + br*x1i + bi*x1r;
                    cre[j1] = dr*x0r - di*x0i + er*x1r - ei*x1i;
                    cim[j1] = dr*x0i + di*x0r + er*x1i + ei*x1r;
                }
            }
            // CNOT ring, range r = (l % 3) + 1 : new[j] = old[j^tb] when control set
            const int r = (l % 3) + 1;
            for (int w = 0; w < 4; ++w) {
                const int cb = 1 << (3 - w);
                const int tb = 1 << (3 - ((w + r) & 3));
                #pragma unroll
                for (int j = 0; j < 16; ++j) {
                    if ((j & cb) && !(j & tb)) {
                        const int k = j | tb;
                        float tr = cre[j], ti = cim[j];
                        cre[j] = cre[k]; cim[j] = cim[k];
                        cre[k] = tr;     cim[k] = ti;
                    }
                }
            }
        }
        // fold column phase (-i)^popcount(i)
        const int pc = __popc(i) & 3;
        const float pr = (pc == 0) ? 1.f : (pc == 2) ? -1.f : 0.f;
        const float pi = (pc == 1) ? -1.f : (pc == 3) ? 1.f : 0.f;
        #pragma unroll
        for (int j = 0; j < 16; ++j) {
            const float wr = cre[j]*pr - cim[j]*pi;
            const float wi = cre[j]*pi + cim[j]*pr;
            wre[j*16 + i] = (_Float16)wr;
            wim[j*16 + i] = (_Float16)wi;
        }
    }
    // G^T[c][j] = sum_w sign_w(j) * fc_w[c][w],  sign_w(j) = (bit(3-w) of j) ? -1 : +1
    for (int idx = tid; idx < 256; idx += blockDim.x) {
        const int c = idx >> 4, j = idx & 15;
        float g = 0.f;
        if (c < 10) {
            #pragma unroll
            for (int w = 0; w < 4; ++w)
                g += (((j >> (3 - w)) & 1) ? -1.f : 1.f) * fc_w[c*4 + w];
        }
        gt[c*16 + j] = (_Float16)g;
    }
    if (tid < 16) bpad[tid] = (tid < 10) ? fc_b[tid] : 0.f;
}

// ---------------------------------------------------------------------------
// Helpers for the main kernel
// ---------------------------------------------------------------------------
__device__ __forceinline__ V16H make_bm(const float4 xv, const int kh) {
    // Tensor-product amplitudes m_i = prod_w (cos|sin)(x_w/2); wire0 bit = kh.
    const float c0 = __cosf(0.5f*xv.x), s0 = __sinf(0.5f*xv.x);
    const float c1 = __cosf(0.5f*xv.y), s1 = __sinf(0.5f*xv.y);
    const float c2 = __cosf(0.5f*xv.z), s2 = __sinf(0.5f*xv.z);
    const float c3 = __cosf(0.5f*xv.w), s3 = __sinf(0.5f*xv.w);
    const float f0 = kh ? s0 : c0;
    V16H bm;
    #pragma unroll
    for (int v = 0; v < 8; ++v) {
        const float f1 = (v & 4) ? s1 : c1;
        const float f2 = (v & 2) ? s2 : c2;
        const float f3 = (v & 1) ? s3 : c3;
        bm.e[v] = (_Float16)(f0 * f1 * f2 * f3);
    }
    #pragma unroll
    for (int v = 4; v < 8; ++v) bm.u[v] = 0u;
    return bm;
}

__device__ __forceinline__ V16H make_p(const v8f dre, const v8f dim) {
    V16H ap;
    #pragma unroll
    for (int v = 0; v < 8; ++v)
        ap.e[v] = (_Float16)(dre[v]*dre[v] + dim[v]*dim[v]);
    #pragma unroll
    for (int v = 4; v < 8; ++v) ap.u[v] = 0u;
    return ap;
}

__device__ __forceinline__ void store_tile(float* __restrict__ out, const int B,
                                           const int base, const int s,
                                           const int kh, const v8f d) {
    // D3: lane -> channel c = s, vgpr v -> sample base + 8*kh + v
    if (base + 16 <= B) {
        // full tile: one address computation, stores use immediate offsets
        if (s < 10) {
            float* p = out + (size_t)(base + kh*8)*10 + s;
            #pragma unroll
            for (int v = 0; v < 8; ++v) p[v*10] = d[v];
        }
    } else if (base < B) {
        // partial tail tile
        if (s < 10) {
            #pragma unroll
            for (int v = 0; v < 8; ++v) {
                const int smp = base + kh*8 + v;
                if (smp < B) out[(size_t)smp*10 + s] = d[v];
            }
        }
    }
}

// ---------------------------------------------------------------------------
// Main kernel: one wave processes 32 samples (2 independent 16-sample tiles)
// per iteration; the two WMMA chains interleave to hide WMMA->VALU hazards.
// Per tile:
//   D1[j,s] = W're[j,:] . m_s   D2[j,s] = W'im[j,:] . m_s   (K=16 padded to 32)
//   p = D1^2 + D2^2  (C/D layout == A layout for next GEMM -> in-lane cvt only)
//   D3[s,c] = p[s,:] . G[:,c] + fc_b[c]
// ---------------------------------------------------------------------------
__global__ void __launch_bounds__(256)
qnn_main(const float* __restrict__ x, int B, int numPairs,
         const _Float16* __restrict__ wre,
         const _Float16* __restrict__ wim,
         const _Float16* __restrict__ gt,
         const float* __restrict__ bpad,
         float* __restrict__ out)
{
    const int lane   = threadIdx.x & 31;
    const int wid    = blockIdx.x * (blockDim.x >> 5) + (threadIdx.x >> 5);
    const int nwaves = gridDim.x * (blockDim.x >> 5);
    const int s  = lane & 15;       // row/col/sample index within tile
    const int kh = lane >> 4;       // K-half select

    // Constant WMMA operands: one b128 load each (8 contiguous halves), high half zero.
    V16H aRe, aIm, bG;
    {
        const uint4 r0 = *(const uint4*)(wre + s*16 + kh*8);
        const uint4 r1 = *(const uint4*)(wim + s*16 + kh*8);
        const uint4 r2 = *(const uint4*)(gt  + s*16 + kh*8);
        aRe.u[0]=r0.x; aRe.u[1]=r0.y; aRe.u[2]=r0.z; aRe.u[3]=r0.w;
        aIm.u[0]=r1.x; aIm.u[1]=r1.y; aIm.u[2]=r1.z; aIm.u[3]=r1.w;
        bG .u[0]=r2.x; bG .u[1]=r2.y; bG .u[2]=r2.z; bG .u[3]=r2.w;
        #pragma unroll
        for (int v = 4; v < 8; ++v) { aRe.u[v]=0u; aIm.u[v]=0u; bG.u[v]=0u; }
    }
    const float myb = bpad[s];
    const v8f cbias = {myb, myb, myb, myb, myb, myb, myb, myb};
    const v8f zero  = {0.f, 0.f, 0.f, 0.f, 0.f, 0.f, 0.f, 0.f};

    for (int pair = wid; pair < numPairs; pair += nwaves) {
        const int base0 = pair << 5;
        const int base1 = base0 + 16;

        const int i0 = base0 + s;
        const int i1 = base1 + s;
        const float4 xv0 = *(const float4*)(x + (size_t)((i0 < B) ? i0 : (B-1)) * 4);
        const float4 xv1 = *(const float4*)(x + (size_t)((i1 < B) ? i1 : (B-1)) * 4);

        // prefetch next iteration's 512B x slice (one lane per sample)
        {
            const int nb = (pair + nwaves) << 5;
            const int ni = nb + lane;
            if (nb < B)
                __builtin_prefetch(x + (size_t)((ni < B) ? ni : (B-1)) * 4, 0, 3);
        }

        const V16H bm0 = make_bm(xv0, kh);
        const V16H bm1 = make_bm(xv1, kh);

        // two independent WMMA chains (scheduler interleaves to hide hazards)
        const v8f re0 = __builtin_amdgcn_wmma_f32_16x16x32_f16(
            false, aRe.v, false, bm0.v, (short)0, zero, false, false);
        const v8f re1 = __builtin_amdgcn_wmma_f32_16x16x32_f16(
            false, aRe.v, false, bm1.v, (short)0, zero, false, false);
        const v8f im0 = __builtin_amdgcn_wmma_f32_16x16x32_f16(
            false, aIm.v, false, bm0.v, (short)0, zero, false, false);
        const v8f im1 = __builtin_amdgcn_wmma_f32_16x16x32_f16(
            false, aIm.v, false, bm1.v, (short)0, zero, false, false);

        const V16H p0 = make_p(re0, im0);
        const V16H p1 = make_p(re1, im1);

        const v8f o0 = __builtin_amdgcn_wmma_f32_16x16x32_f16(
            false, p0.v, false, bG.v, (short)0, cbias, false, false);
        const v8f o1 = __builtin_amdgcn_wmma_f32_16x16x32_f16(
            false, p1.v, false, bG.v, (short)0, cbias, false, false);

        store_tile(out, B, base0, s, kh, o0);
        store_tile(out, B, base1, s, kh, o1);
    }
}

extern "C" void kernel_launch(void* const* d_in, const int* in_sizes, int n_in,
                              void* d_out, int out_size, void* d_ws, size_t ws_size,
                              hipStream_t stream) {
    const float* x       = (const float*)d_in[0];   // [B,4]
    const float* weights = (const float*)d_in[1];   // [4,4,3]
    const float* fc_w    = (const float*)d_in[2];   // [10,4]
    const float* fc_b    = (const float*)d_in[3];   // [10]
    const int B = in_sizes[0] / 4;

    _Float16* wre  = (_Float16*)d_ws;       // 256 halves
    _Float16* wim  = wre + 256;             // 256 halves
    _Float16* gt   = wim + 256;             // 256 halves
    float*    bpad = (float*)(gt + 256);    // 16 floats

    qnn_setup<<<1, 32, 0, stream>>>(weights, fc_w, fc_b, wre, wim, gt, bpad);

    const int numPairs = (B + 31) / 32;     // 32 samples per wave-iteration
    int blocks = 1024;                      // 8192 waves, ~4 pairs each @ B=1M
    const int wavesPerBlock = 256 / 32;
    if (blocks * wavesPerBlock > numPairs)
        blocks = (numPairs + wavesPerBlock - 1) / wavesPerBlock;
    if (blocks < 1) blocks = 1;

    qnn_main<<<blocks, 256, 0, stream>>>(x, B, numPairs, wre, wim, gt, bpad,
                                         (float*)d_out);
}
